// RNNEncoder_78434692760286
// MI455X (gfx1250) — compile-verified, matrix operand order checked by reading, not run on hardware
//
#include <hip/hip_runtime.h>
#include <hip/hip_bf16.h>

// ---------------------------------------------------------------------------
// Persistent-WMMA LSTM for MI455X (gfx1250, wave32).
//
//   B=256, T=512, IN=130 (last 2 dropped -> 128), H=512, gates=4H=2048
//   Combined K = 128 (x_t) + 512 (h) = 640  -> 20 WMMA K-tiles of 32.
//
// Grid: 32 persistent workgroups x 256 threads (8 waves).
//   WG wg owns h-columns [wg*16, wg*16+16): it computes gate column tiles
//   at n = g*512 + wg*16 for g in {i,f,g,o}, for all 16 batch row-tiles.
//   Wave w handles row-tiles {2w, 2w+1}; all 4 gate accumulators for a tile
//   live in the same lane slots -> LSTM cell update is register-local.
//   c-state lives in VGPRs for the whole 512-step loop.
//
//   Per step: one grid-wide barrier; h ping-pongs through a global bf16
//   "xh" buffer [256 x 640] (x part refreshed for t+1 during the update).
//
// Weights: packed to bf16 WMMA B-fragments once per call (pack kernel);
//   each WG's 80KB slice is staged into LDS via TDM tensor_load_to_lds
//   (fallback: manual b128 copy).  Output gather hs[len[b]-1] is fused
//   into the update phase.
// ---------------------------------------------------------------------------

#define B_    256
#define T_    512
#define INF_  130
#define INX_  128
#define H_    512
#define KTOT_ 640
#define NKT_  20
#define NWG_  32
#define NTHR_ 256

typedef __bf16 bf16_t;
typedef __attribute__((ext_vector_type(16))) __bf16    v16bf_t;
typedef __attribute__((ext_vector_type(8)))  float     v8f_t;
typedef __attribute__((ext_vector_type(4)))  unsigned  v4u_t;
typedef __attribute__((ext_vector_type(4)))  int       v4i_t;
typedef __attribute__((ext_vector_type(8)))  int       v8i_t;

// bytes per WG slice of packed weights: 4 gates * 20 kt * 32 lanes * 16 bf16
#define WGB_ELEMS (4 * NKT_ * 32 * 16)   // 40960 bf16 = 81920 bytes

__device__ __forceinline__ float sig_(float x) {
  return 1.0f / (1.0f + __expf(-x));
}
__device__ __forceinline__ float tanh_(float x) {
  float e = __expf(2.0f * x);
  return (e - 1.0f) / (e + 1.0f);
}

__device__ __forceinline__ void grid_sync_(unsigned* cnt, unsigned target) {
  __threadfence();
  __syncthreads();
  if (threadIdx.x == 0) {
    __hip_atomic_fetch_add(cnt, 1u, __ATOMIC_RELEASE, __HIP_MEMORY_SCOPE_AGENT);
    while (__hip_atomic_load(cnt, __ATOMIC_ACQUIRE, __HIP_MEMORY_SCOPE_AGENT) < target) {
      __builtin_amdgcn_s_sleep(1);
    }
  }
  __syncthreads();
}

// ---------------------------------------------------------------------------
// Pack W_ih (2048x128) ++ W_hh (2048x512) into bf16 B-fragments:
//   flat e = (((wg*4 + g)*20 + kt)*32 + lane)*16 + i
//   n = g*512 + wg*16 + (lane&15);  k = kt*32 + (lane>>4)*16 + i
// Each WG's slice is contiguous (80KB) so it can be TDM'd to LDS in one op,
// and every per-lane fragment load in the main kernel is 32B contiguous.
// ---------------------------------------------------------------------------
__global__ void pack_w_kernel(const float* __restrict__ Wih,
                              const float* __restrict__ Whh,
                              bf16_t* __restrict__ wsB) {
  int e = blockIdx.x * NTHR_ + threadIdx.x;      // < 2048*640 = 1310720
  int i    = e & 15;
  int lane = (e >> 4) & 31;
  int rest = e >> 9;                              // (wg*4+g)*20 + kt
  int kt   = rest % NKT_;
  int gwg  = rest / NKT_;
  int g    = gwg & 3;
  int wg   = gwg >> 2;
  int n = g * H_ + wg * 16 + (lane & 15);
  int k = kt * 32 + (lane >> 4) * 16 + i;
  float v = (k < INX_) ? Wih[(size_t)n * INX_ + k]
                       : Whh[(size_t)n * H_ + (k - INX_)];
  wsB[e] = (bf16_t)v;
}

__global__ void pack_bias_kernel(const float* __restrict__ bih,
                                 const float* __restrict__ bhh,
                                 float* __restrict__ bias) {
  int e = blockIdx.x * NTHR_ + threadIdx.x;
  if (e < 4 * H_) bias[e] = bih[e] + bhh[e];
}

// ---------------------------------------------------------------------------
// Persistent LSTM kernel.
// ---------------------------------------------------------------------------
__global__ __launch_bounds__(NTHR_, 1)
void lstm_persist_kernel(const float* __restrict__ inputs,
                         const int*   __restrict__ ilen,
                         const float* __restrict__ h0,
                         const float* __restrict__ c0,
                         const bf16_t* __restrict__ wsB,
                         const float* __restrict__ bias,
                         bf16_t* __restrict__ xh0,
                         bf16_t* __restrict__ xh1,
                         unsigned* __restrict__ syncp,
                         float* __restrict__ out) {
  extern __shared__ char smem_raw[];              // 80KB dynamic LDS: B frags
  const v16bf_t* ldsB = (const v16bf_t*)smem_raw;

  const int tid  = threadIdx.x;
  const int wave = tid >> 5;         // 0..7
  const int lane = tid & 31;
  const int half = lane >> 4;        // 0/1 -> C rows M and M+8
  const int ln   = lane & 15;        // C column within tile
  const int wg   = blockIdx.x;       // 0..31
  const int colg = wg * 16 + ln;     // owned h column

  // ---- stage this WG's 80KB packed-weight slice into LDS -----------------
#if defined(__gfx1250__) && __has_builtin(__builtin_amdgcn_tensor_load_to_lds)
  if (tid < 32) {                    // wave 0 issues one TDM op
    unsigned long long ga =
        (unsigned long long)(uintptr_t)(wsB + (size_t)wg * WGB_ELEMS);
    // D# group0: count=1, lds_addr=0 (dynamic LDS base), global addr, type=2
    v4u_t g0;
    g0[0] = 1u;
    g0[1] = 0u;
    g0[2] = (unsigned)ga;
    g0[3] = (unsigned)((ga >> 32) & 0x01FFFFFFu) | 0x80000000u;
    // D# group1: data_size=8B (code 3); 1-D tensor of 10240 8B units,
    // tile_dim0=10240, dim1 = 1, strides = 10240.
    v8i_t g1;
    g1[0] = 0x00030000;              // data_size=3 (8 bytes)
    g1[1] = (int)(10240u << 16);     // tensor_dim0 lo16 @ bit48
    g1[2] = (int)(1u << 16);         // tensor_dim0 hi=0 | tensor_dim1=1
    g1[3] = (int)(10240u << 16);     // tensor_dim1 hi=0 | tile_dim0=10240
    g1[4] = 1;                       // tile_dim1=1, tile_dim2=0
    g1[5] = 10240;                   // tensor_dim0_stride lo32
    g1[6] = (int)(10240u << 16);     // stride0 hi=0 | tensor_dim1_stride lo16
    g1[7] = 0;
    v4i_t zz = {0, 0, 0, 0};
#if __clang_major__ >= 23
    v8i_t zz8 = {0, 0, 0, 0, 0, 0, 0, 0};
    __builtin_amdgcn_tensor_load_to_lds(g0, g1, zz, zz, zz8, 0);
#else
    __builtin_amdgcn_tensor_load_to_lds(g0, g1, zz, zz, 0);
#endif
#if __has_builtin(__builtin_amdgcn_s_wait_tensorcnt)
    __builtin_amdgcn_s_wait_tensorcnt(0);
#endif
  }
#else
  {
    const uint4* gsrc = (const uint4*)(wsB + (size_t)wg * WGB_ELEMS);
    uint4* ldst = (uint4*)smem_raw;
    for (int i = tid; i < (WGB_ELEMS * 2) / 16; i += NTHR_) ldst[i] = gsrc[i];
  }
#endif
  __syncthreads();

  // ---- per-thread persistent state ---------------------------------------
  v8f_t cst[2];                      // c-state for (bt, r)
  int   lm1[16];                     // len-1 per owned row
  float bg[4];                       // combined bias per gate at colg
#pragma unroll
  for (int g = 0; g < 4; ++g) bg[g] = bias[g * H_ + colg];

#pragma unroll
  for (int bt = 0; bt < 2; ++bt) {
    const int btile = wave * 2 + bt;
#pragma unroll
    for (int r = 0; r < 8; ++r) {
      const int row = btile * 16 + half * 8 + r;
      cst[bt][r]    = c0[(size_t)row * H_ + colg];
      lm1[bt * 8 + r] = ilen[row] - 1;
      xh0[(size_t)row * KTOT_ + INX_ + colg] = (bf16_t)h0[(size_t)row * H_ + colg];
    }
  }
  // x-part for t=0: WG wg fills rows [wg*8, wg*8+8), 4 features per lane
  {
    const int brow = wg * 8 + wave;
    const int k0 = lane * 4;
    const float* s = inputs + (size_t)brow * T_ * INF_ + k0;
    float2 p0 = *(const float2*)s;
    float2 p1 = *(const float2*)(s + 2);
    union { bf16_t h[4]; uint2 u; } pk;
    pk.h[0] = (bf16_t)p0.x; pk.h[1] = (bf16_t)p0.y;
    pk.h[2] = (bf16_t)p1.x; pk.h[3] = (bf16_t)p1.y;
    *(uint2*)(xh0 + (size_t)brow * KTOT_ + k0) = pk.u;
  }

  unsigned tgt = NWG_;
  grid_sync_(syncp, tgt); tgt += NWG_;

  // ---- time loop ---------------------------------------------------------
  for (int t = 0; t < T_; ++t) {
    const bf16_t* xh  = (t & 1) ? xh1 : xh0;
    bf16_t*       xhn = (t & 1) ? xh0 : xh1;

#pragma unroll
    for (int bt = 0; bt < 2; ++bt) {
      const int btile = wave * 2 + bt;
      v8f_t a0 = {0,0,0,0,0,0,0,0};
      v8f_t a1 = {0,0,0,0,0,0,0,0};
      v8f_t a2 = {0,0,0,0,0,0,0,0};
      v8f_t a3 = {0,0,0,0,0,0,0,0};
      const bf16_t* aptr = xh + (size_t)(btile * 16 + ln) * KTOT_ + half * 16;
#pragma unroll 4
      for (int kt = 0; kt < NKT_; ++kt) {
        v16bf_t af = *(const v16bf_t*)(aptr + kt * 32);      // 32B coalesced
        v16bf_t b0 = ldsB[(0 * NKT_ + kt) * 32 + lane];
        v16bf_t b1 = ldsB[(1 * NKT_ + kt) * 32 + lane];
        v16bf_t b2 = ldsB[(2 * NKT_ + kt) * 32 + lane];
        v16bf_t b3 = ldsB[(3 * NKT_ + kt) * 32 + lane];
        a0 = __builtin_amdgcn_wmma_f32_16x16x32_bf16(false, af, false, b0,
                                                     (short)0, a0, false, false);
        a1 = __builtin_amdgcn_wmma_f32_16x16x32_bf16(false, af, false, b1,
                                                     (short)0, a1, false, false);
        a2 = __builtin_amdgcn_wmma_f32_16x16x32_bf16(false, af, false, b2,
                                                     (short)0, a2, false, false);
        a3 = __builtin_amdgcn_wmma_f32_16x16x32_bf16(false, af, false, b3,
                                                     (short)0, a3, false, false);
      }
      // register-local LSTM cell update (i,f,g,o share lane slots)
#pragma unroll
      for (int r = 0; r < 8; ++r) {
        float iv = sig_(a0[r] + bg[0]);
        float fv = sig_(a1[r] + bg[1]);
        float gv = tanh_(a2[r] + bg[2]);
        float ov = sig_(a3[r] + bg[3]);
        float c  = fv * cst[bt][r] + iv * gv;
        cst[bt][r] = c;
        float h  = ov * tanh_(c);
        const int row = btile * 16 + half * 8 + r;
        xhn[(size_t)row * KTOT_ + INX_ + colg] = (bf16_t)h;
        if (lm1[bt * 8 + r] == t) out[(size_t)row * H_ + colg] = h;
      }
    }

    // refresh x-part of next buffer for step t+1
    if (t + 1 < T_) {
      const int brow = wg * 8 + wave;
      const int k0 = lane * 4;
      const float* s = inputs + ((size_t)brow * T_ + (t + 1)) * INF_ + k0;
      float2 p0 = *(const float2*)s;
      float2 p1 = *(const float2*)(s + 2);
      union { bf16_t h[4]; uint2 u; } pk;
      pk.h[0] = (bf16_t)p0.x; pk.h[1] = (bf16_t)p0.y;
      pk.h[2] = (bf16_t)p1.x; pk.h[3] = (bf16_t)p1.y;
      *(uint2*)(xhn + (size_t)brow * KTOT_ + k0) = pk.u;
      if (t + 2 < T_)
        __builtin_prefetch(inputs + ((size_t)brow * T_ + (t + 2)) * INF_ + k0, 0, 1);
    }

    grid_sync_(syncp, tgt); tgt += NWG_;
  }
}

// ---------------------------------------------------------------------------
// Workspace layout (bytes):
//   [0,        2621440)  packed bf16 weights (2048*640)
//   [2621440,  2629632)  combined bias f32[2048]
//   [2629632,  2957312)  xh ping  bf16[256*640]
//   [2957312,  3284992)  xh pong  bf16[256*640]
//   [3284992,  3285248)  grid-sync counter
// ---------------------------------------------------------------------------
extern "C" void kernel_launch(void* const* d_in, const int* in_sizes, int n_in,
                              void* d_out, int out_size, void* d_ws, size_t ws_size,
                              hipStream_t stream) {
  (void)in_sizes; (void)n_in; (void)out_size; (void)ws_size;
  const float* inputs = (const float*)d_in[0];
  const int*   ilen   = (const int*)d_in[1];
  const float* h0     = (const float*)d_in[2];
  const float* c0     = (const float*)d_in[3];
  const float* Wih    = (const float*)d_in[4];
  const float* Whh    = (const float*)d_in[5];
  const float* bih    = (const float*)d_in[6];
  const float* bhh    = (const float*)d_in[7];
  float* out = (float*)d_out;

  char* ws = (char*)d_ws;
  bf16_t*   wsB   = (bf16_t*)ws;
  float*    bias  = (float*)(ws + 2621440);
  bf16_t*   xh0   = (bf16_t*)(ws + 2629632);
  bf16_t*   xh1   = (bf16_t*)(ws + 2957312);
  unsigned* syncp = (unsigned*)(ws + 3284992);

  hipMemsetAsync(syncp, 0, 256, stream);
  pack_w_kernel<<<dim3((2048 * 640) / NTHR_), dim3(NTHR_), 0, stream>>>(Wih, Whh, wsB);
  pack_bias_kernel<<<dim3(8), dim3(NTHR_), 0, stream>>>(bih, bhh, bias);
  lstm_persist_kernel<<<dim3(NWG_), dim3(NTHR_), WGB_ELEMS * 2, stream>>>(
      inputs, ilen, h0, c0, wsB, bias, xh0, xh1, syncp, out);
}
